// XLAttentionLayer_51788715655221
// MI455X (gfx1250) — matrix operand order repaired and checked by reading, not compiled, not run on hardware
//
#include <hip/hip_runtime.h>
#include <hip/hip_bf16.h>

// ---------------------------------------------------------------------------
// Transformer-XL layer forward for MI455X (gfx1250, wave32, WMMA bf16,
// double-buffered async global->LDS staging)
// ---------------------------------------------------------------------------

typedef __bf16 bf16_t;
typedef __attribute__((ext_vector_type(16))) __bf16 v16bf;
typedef __attribute__((ext_vector_type(8)))  float  v8f;

#define XB 4
#define XT 2048
#define XM 2048
#define XK 4096
#define XD 1024
#define XH 16
#define XDH 64
#define XF 4096

// ------------------------- CDNA5 async LDS staging -------------------------
// GLOBAL_LOAD_ASYNC_TO_LDS_B128: per-lane 16B global->LDS DMA, ASYNCcnt.
// Flat LDS addresses carry the LDS byte offset in the low 32 bits (ISA 10.2).
// ASYNCcnt retires in order -> s_wait_asynccnt N pipelines double buffering.
__device__ __forceinline__ void async_b128(void* lds_dst, const void* gsrc) {
  unsigned loff = (unsigned)(uintptr_t)lds_dst;
  asm volatile("global_load_async_to_lds_b128 %0, %1, off"
               :: "v"(loff), "v"(gsrc)
               : "memory");
}
__device__ __forceinline__ void async_wait0()  { asm volatile("s_wait_asynccnt 0"  ::: "memory"); }
__device__ __forceinline__ void async_wait8()  { asm volatile("s_wait_asynccnt 8"  ::: "memory"); }
__device__ __forceinline__ void async_wait16() { asm volatile("s_wait_asynccnt 16" ::: "memory"); }

// ------------------------------ WMMA helpers -------------------------------

__device__ __forceinline__ v8f wmma_bf16(v16bf a, v16bf b, v8f c) {
  return __builtin_amdgcn_wmma_f32_16x16x32_bf16(
      /*neg_a=*/false, a, /*neg_b=*/false, b,
      /*c_mod=*/(short)0, c, /*reuse_a=*/false, /*reuse_b=*/false);
}

// A fragment: 16x32 bf16 (rows m, cols k), row-major source, stride ld.
// Per lane: two contiguous 8-element (16B) chunks -> ds_load_b128.
__device__ __forceinline__ v16bf load_a_frag(const bf16_t* a, int ld, int kbase, int lane) {
  const int halfl = lane >> 4;
  const int m = lane & 15;
  v16bf r;
#pragma unroll
  for (int e = 0; e < 16; ++e) {
    const int vg = e >> 1, pp = e & 1;
    const int k = ((vg < 4) ? (2 * vg) : (16 + 2 * (vg - 4))) + halfl * 8 + pp;
    r[e] = a[m * ld + kbase + k];
  }
  return r;
}

// B fragment B[k][n] sourced from transposed storage src[n][k] (row-major n x k).
// Per lane: two contiguous 8-element k chunks (16B each, same k order as A) -> ds_load_b128.
__device__ __forceinline__ v16bf load_bT_frag(const bf16_t* src, int ld, int kbase, int lane) {
  const int halfl = lane >> 4;
  const int n = lane & 15;
  v16bf r;
#pragma unroll
  for (int e = 0; e < 16; ++e) {
    const int vg = e >> 1, pp = e & 1;
    const int k = ((vg < 4) ? (2 * vg) : (16 + 2 * (vg - 4))) + halfl * 8 + pp;
    r[e] = src[n * ld + kbase + k];
  }
  return r;
}

__device__ __forceinline__ float gelu_tanh(float x) {
  const float c = 0.7978845608028654f;  // sqrt(2/pi)
  float t = tanhf(c * (x + 0.044715f * x * x * x));
  return 0.5f * x * (1.0f + t);
}

// ------------------------------ small kernels ------------------------------

__global__ void xl_cvt_f32_bf16(const float* __restrict__ in, bf16_t* __restrict__ out,
                                size_t n) {
  size_t i = (size_t)blockIdx.x * blockDim.x + threadIdx.x;
  if (i < n) out[i] = (bf16_t)in[i];
}

// in f32 [R][C] -> out bf16 [C][R]
__global__ void xl_cvt_T(const float* __restrict__ in, bf16_t* __restrict__ out,
                         int R, int C) {
  size_t i = (size_t)blockIdx.x * blockDim.x + threadIdx.x;
  if (i >= (size_t)R * C) return;
  int r = (int)(i / C), c = (int)(i - (size_t)r * C);
  out[(size_t)c * R + r] = (bf16_t)in[i];
}

__global__ void xl_concat_cvt(const float* __restrict__ mem, const float* __restrict__ x,
                              bf16_t* __restrict__ c) {
  size_t i = (size_t)blockIdx.x * blockDim.x + threadIdx.x;
  if (i >= (size_t)XB * XK * XD) return;
  size_t b = i / ((size_t)XK * XD);
  size_t rem = i - b * (size_t)XK * XD;
  int j = (int)(rem >> 10);
  int d = (int)(rem & (XD - 1));
  float v = (j < XM) ? mem[(b * XM + j) * XD + d]
                     : x[(b * XT + (j - XM)) * XD + d];
  c[i] = (bf16_t)v;
}

__global__ void xl_posemb(bf16_t* __restrict__ out) {
  size_t i = (size_t)blockIdx.x * blockDim.x + threadIdx.x;
  if (i >= (size_t)XK * XD) return;
  int jj = (int)(i >> 10);
  int d = (int)(i & (XD - 1));
  int f = (d < XD / 2) ? d : (d - XD / 2);
  float inv = powf(10000.0f, -((float)(2 * f)) * (1.0f / (float)XD));
  float ang = (float)(XK - 1 - jj) * inv;
  out[i] = (bf16_t)((d < XD / 2) ? sinf(ang) : cosf(ang));
}

// q f32 [B*T, D] -> qu/qv bf16 [B][H][T][DH] with u/v added per head
__global__ void xl_prep_q(const float* __restrict__ q, const float* __restrict__ u,
                          const float* __restrict__ v, bf16_t* __restrict__ qu,
                          bf16_t* __restrict__ qv) {
  size_t i = (size_t)blockIdx.x * blockDim.x + threadIdx.x;
  if (i >= (size_t)XB * XT * XD) return;
  size_t b = i / ((size_t)XT * XD);
  size_t rem = i - b * (size_t)XT * XD;
  int t = (int)(rem >> 10);
  int dc = (int)(rem & (XD - 1));
  int h = dc >> 6, d = dc & 63;
  float qq = q[i];
  size_t o = (((b * XH + h) * XT) + t) * XDH + d;
  qu[o] = (bf16_t)(qq + u[h * XDH + d]);
  qv[o] = (bf16_t)(qq + v[h * XDH + d]);
}

// in f32 [B*K, D] -> out bf16 [B][H][K][DH]   (K matrix)
__global__ void xl_prep_kv(const float* __restrict__ in, bf16_t* __restrict__ out) {
  size_t i = (size_t)blockIdx.x * blockDim.x + threadIdx.x;
  if (i >= (size_t)XB * XK * XD) return;
  size_t b = i / ((size_t)XK * XD);
  size_t rem = i - b * (size_t)XK * XD;
  int j = (int)(rem >> 10);
  int dc = (int)(rem & (XD - 1));
  int h = dc >> 6, d = dc & 63;
  out[(((b * XH + h) * XK) + j) * XDH + d] = (bf16_t)in[i];
}

// in f32 [B*K, D] -> out bf16 [B][H][DH][K]   (V transposed for K-contiguous PV)
__global__ void xl_prep_vT(const float* __restrict__ in, bf16_t* __restrict__ out) {
  size_t i = (size_t)blockIdx.x * blockDim.x + threadIdx.x;
  if (i >= (size_t)XB * XK * XD) return;
  size_t b = i / ((size_t)XK * XD);
  size_t rem = i - b * (size_t)XK * XD;
  int j = (int)(rem >> 10);
  int dc = (int)(rem & (XD - 1));
  int h = dc >> 6, d = dc & 63;
  out[(((b * XH + h) * XDH) + d) * XK + j] = (bf16_t)in[i];
}

// r f32 [K, D] -> out bf16 [H][K][DH]
__global__ void xl_prep_r(const float* __restrict__ in, bf16_t* __restrict__ out) {
  size_t i = (size_t)blockIdx.x * blockDim.x + threadIdx.x;
  if (i >= (size_t)XK * XD) return;
  int jj = (int)(i >> 10);
  int dc = (int)(i & (XD - 1));
  int h = dc >> 6, d = dc & 63;
  out[(((size_t)h * XK) + jj) * XDH + d] = (bf16_t)in[i];
}

// Residual + LayerNorm over last dim (1024): out = LN(a + res) * g + b
__global__ __launch_bounds__(256) void xl_ln_res(const float* __restrict__ a,
                                                 const float* __restrict__ res,
                                                 const float* __restrict__ g,
                                                 const float* __restrict__ be,
                                                 float* __restrict__ out) {
  __shared__ float r1[256], r2[256];
  const int row = blockIdx.x;
  const int t = threadIdx.x;
  const float* pa = a + (size_t)row * XD;
  const float* pr = res + (size_t)row * XD;
  float v[4], s = 0.f, s2 = 0.f;
#pragma unroll
  for (int i = 0; i < 4; ++i) {
    float x = pa[t + i * 256] + pr[t + i * 256];
    v[i] = x; s += x; s2 += x * x;
  }
  r1[t] = s; r2[t] = s2;
  __syncthreads();
  for (int off = 128; off > 0; off >>= 1) {
    if (t < off) { r1[t] += r1[t + off]; r2[t] += r2[t + off]; }
    __syncthreads();
  }
  float mu = r1[0] * (1.0f / XD);
  float var = r2[0] * (1.0f / XD) - mu * mu;
  float rstd = rsqrtf(var + 1e-5f);
#pragma unroll
  for (int i = 0; i < 4; ++i) {
    int col = t + i * 256;
    out[(size_t)row * XD + col] = (v[i] - mu) * rstd * g[col] + be[col];
  }
}

// --------------------------- WMMA GEMM (generic) ---------------------------
// C[M,N] f32 = A[M,Kd] bf16 @ B[Kd,N] bf16 (+bias)(+gelu). B passed TRANSPOSED
// as BT[N][Kd]. 128 threads / 4 waves, 64x64 tile, K-step 64, double-buffered
// async staging (8 async B128 per thread per tile -> s_wait_asynccnt 8).
__global__ __launch_bounds__(128) void xl_gemm_bf16(
    const bf16_t* __restrict__ A, const bf16_t* __restrict__ BT,
    const float* __restrict__ bias, float* __restrict__ C,
    int N, int Kd, int act) {
  __shared__ __align__(16) bf16_t As[2][64 * 64];
  __shared__ __align__(16) bf16_t Bs[2][64 * 64];  // [n][k]
  const int tid = threadIdx.x;
  const int wave = tid >> 5, lane = tid & 31;
  const int halfl = lane >> 4, l16 = lane & 15;
  const int m0 = blockIdx.y * 64, n0 = blockIdx.x * 64;
  const int wm = (wave >> 1) * 32, wn = (wave & 1) * 32;

  v8f acc[2][2];
#pragma unroll
  for (int i = 0; i < 2; ++i)
#pragma unroll
    for (int j = 0; j < 2; ++j)
#pragma unroll
      for (int r = 0; r < 8; ++r) acc[i][j][r] = 0.0f;

  auto stage = [&](int kk, int bank) {
#pragma unroll
    for (int it = 0; it < 4; ++it) {
      const int q = tid + it * 128;             // fixed trip count: no exec masking
      const int r = q >> 3, c8 = (q & 7) * 8;
      async_b128(&As[bank][r * 64 + c8], &A[(size_t)(m0 + r) * Kd + kk + c8]);
      async_b128(&Bs[bank][r * 64 + c8], &BT[(size_t)(n0 + r) * Kd + kk + c8]);
    }
  };

  stage(0, 0);
  const int nk = Kd >> 6;
  for (int kt = 0; kt < nk; ++kt) {
    const int bank = kt & 1;
    if (kt + 1 < nk) {
      stage((kt + 1) << 6, bank ^ 1);
      async_wait8();    // in-order retire: tile kt resident, kt+1 in flight
    } else {
      async_wait0();
    }
    __syncthreads();
#pragma unroll
    for (int kc = 0; kc < 2; ++kc) {
      v16bf a0 = load_a_frag(&As[bank][wm * 64], 64, kc * 32, lane);
      v16bf a1 = load_a_frag(&As[bank][(wm + 16) * 64], 64, kc * 32, lane);
      v16bf b0 = load_bT_frag(&Bs[bank][wn * 64], 64, kc * 32, lane);
      v16bf b1 = load_bT_frag(&Bs[bank][(wn + 16) * 64], 64, kc * 32, lane);
      acc[0][0] = wmma_bf16(a0, b0, acc[0][0]);
      acc[0][1] = wmma_bf16(a0, b1, acc[0][1]);
      acc[1][0] = wmma_bf16(a1, b0, acc[1][0]);
      acc[1][1] = wmma_bf16(a1, b1, acc[1][1]);
    }
    __syncthreads();   // protect bank^1 against next stage()
  }

#pragma unroll
  for (int i = 0; i < 2; ++i)
#pragma unroll
    for (int j = 0; j < 2; ++j) {
      const int col = n0 + wn + j * 16 + l16;
      const float bv = bias ? bias[col] : 0.0f;
#pragma unroll
      for (int r = 0; r < 8; ++r) {
        const int m = m0 + wm + i * 16 + r + 8 * halfl;
        float val = acc[i][j][r] + bv;
        if (act == 1) val = gelu_tanh(val);
        C[(size_t)m * N + col] = val;
      }
    }
}

// --------------------------- Flash XL attention ----------------------------
// grid (T/64, H, B), block 128 (4 waves, 16 query rows each)
// S[i,j] = ((q+u)_i . k_j + (q+v)_i . r_{j-i+T-1}) / 8, causal mask,
// online softmax, out += P @ V. Double-buffered async staging of K/V^T/R band
// (16 async B128 per thread per key tile -> s_wait_asynccnt 16).
__global__ __launch_bounds__(128) void xl_flash_attn(
    const bf16_t* __restrict__ qu, const bf16_t* __restrict__ qv,
    const bf16_t* __restrict__ kb, const bf16_t* __restrict__ vbT,
    const bf16_t* __restrict__ rb, const int* __restrict__ is_causal,
    float* __restrict__ out) {
  __shared__ __align__(16) bf16_t Qu[64 * XDH];
  __shared__ __align__(16) bf16_t Qv[64 * XDH];
  __shared__ __align__(16) bf16_t Ks[2][64 * XDH];    // [key][d]
  __shared__ __align__(16) bf16_t VsT[2][XDH * 64];   // [d][key]
  __shared__ __align__(16) bf16_t Rs[2][128 * XDH];   // [band][d]
  __shared__ __align__(16) bf16_t P[64 * 64];         // [row][key]
  __shared__ float SAC[64 * 64];
  __shared__ float BDs[64 * 128];
  __shared__ float mstate[64], lstate[64], alpha_s[64], mnew_s[64];
  __shared__ float rowmax2[128], rowsum2[128];

  const int qi = blockIdx.x, h = blockIdx.y, b = blockIdx.z;
  const int i0 = qi * 64;
  const int tid = threadIdx.x;
  const int wave = tid >> 5, lane = tid & 31;
  const int halfl = lane >> 4, l16 = lane & 15;
  const int r0 = wave * 16;

  const bf16_t* quB = qu + (((size_t)(b * XH + h) * XT) + i0) * XDH;
  const bf16_t* qvB = qv + (((size_t)(b * XH + h) * XT) + i0) * XDH;
  const bf16_t* kbB = kb + ((size_t)(b * XH + h) * XK) * XDH;
  const bf16_t* vtB = vbT + ((size_t)(b * XH + h) * XDH) * XK;
  const bf16_t* rbB = rb + ((size_t)h * XK) * XDH;

  auto stage_kvr = [&](int j0, int bank) {
#pragma unroll
    for (int it = 0; it < 4; ++it) {
      const int q = tid + it * 128;
      async_b128(&Ks[bank][q * 8], kbB + (size_t)j0 * XDH + q * 8);
      const int d = q >> 3, c8 = (q & 7) * 8;
      async_b128(&VsT[bank][d * 64 + c8], vtB + (size_t)d * XK + j0 + c8);
    }
    const int jjbase = j0 - i0 + (XT - 1) - 63;
#pragma unroll
    for (int it = 0; it < 8; ++it) {
      const int q = tid + it * 128;
      const int rr = q >> 3, c8 = (q & 7) * 8;
      int jj = jjbase + rr;
      jj = jj < 0 ? 0 : (jj > XK - 1 ? XK - 1 : jj);
      async_b128(&Rs[bank][rr * XDH + c8], rbB + (size_t)jj * XDH + c8);
    }
  };

  // Q tiles (contiguous 64 rows) + first key tile
#pragma unroll
  for (int it = 0; it < 4; ++it) {
    const int q = tid + it * 128;
    async_b128(&Qu[q * 8], quB + (size_t)q * 8);
    async_b128(&Qv[q * 8], qvB + (size_t)q * 8);
  }
  stage_kvr(0, 0);
  if (tid < 64) { mstate[tid] = -1e30f; lstate[tid] = 0.0f; }

  v8f oacc[4];
#pragma unroll
  for (int nt = 0; nt < 4; ++nt)
#pragma unroll
    for (int r = 0; r < 8; ++r) oacc[nt][r] = 0.0f;

  const int causal = is_causal[0];
  const int ntiles = causal ? (qi + 33) : (XK / 64);

  for (int jt = 0; jt < ntiles; ++jt) {
    const int j0 = jt * 64;
    const int bank = jt & 1;
    if (jt + 1 < ntiles) {
      stage_kvr((jt + 1) * 64, bank ^ 1);
      async_wait16();   // tile jt (and Q) resident; jt+1 streams during compute
    } else {
      async_wait0();
    }
    __syncthreads();

    // AC = (q+u) @ K^T -> SAC (per-wave 16 rows x 64 cols)
#pragma unroll
    for (int nt = 0; nt < 4; ++nt) {
      v8f a8;
#pragma unroll
      for (int r = 0; r < 8; ++r) a8[r] = 0.0f;
#pragma unroll
      for (int kc = 0; kc < 2; ++kc) {
        v16bf af = load_a_frag(Qu + r0 * XDH, XDH, kc * 32, lane);
        v16bf bt = load_bT_frag(&Ks[bank][(nt * 16) * XDH], XDH, kc * 32, lane);
        a8 = wmma_bf16(af, bt, a8);
      }
#pragma unroll
      for (int r = 0; r < 8; ++r)
        SAC[(r0 + r + 8 * halfl) * 64 + nt * 16 + l16] = a8[r];
    }
    // BD band = (q+v) @ Rband^T -> BDs (16 rows x 128 cols per wave)
#pragma unroll
    for (int nt = 0; nt < 8; ++nt) {
      v8f a8;
#pragma unroll
      for (int r = 0; r < 8; ++r) a8[r] = 0.0f;
#pragma unroll
      for (int kc = 0; kc < 2; ++kc) {
        v16bf af = load_a_frag(Qv + r0 * XDH, XDH, kc * 32, lane);
        v16bf bt = load_bT_frag(&Rs[bank][(nt * 16) * XDH], XDH, kc * 32, lane);
        a8 = wmma_bf16(af, bt, a8);
      }
#pragma unroll
      for (int r = 0; r < 8; ++r)
        BDs[(r0 + r + 8 * halfl) * 128 + nt * 16 + l16] = a8[r];
    }
    __syncthreads();

    // online softmax: 2 threads per row, 32 cols each
    const int row = tid >> 1, ch = tid & 1;
    float mloc = -1e30f;
#pragma unroll 4
    for (int j = ch * 32; j < ch * 32 + 32; ++j) {
      float s = (SAC[row * 64 + j] + BDs[row * 128 + (j - row + 63)]) * 0.125f;
      if (causal && (j0 + j) > (i0 + row + XM)) s = -1e30f;
      SAC[row * 64 + j] = s;
      mloc = fmaxf(mloc, s);
    }
    rowmax2[row * 2 + ch] = mloc;
    __syncthreads();
    if (ch == 0) {
      float mt = fmaxf(rowmax2[row * 2], rowmax2[row * 2 + 1]);
      float mold = mstate[row];
      float mn = fmaxf(mold, mt);
      mnew_s[row] = mn;
      alpha_s[row] = __expf(mold - mn);
    }
    __syncthreads();
    {
      const float mn = mnew_s[row];
      float ssum = 0.0f;
#pragma unroll 4
      for (int j = ch * 32; j < ch * 32 + 32; ++j) {
        float pv = __expf(SAC[row * 64 + j] - mn);
        P[row * 64 + j] = (bf16_t)pv;
        ssum += pv;
      }
      rowsum2[row * 2 + ch] = ssum;
    }
    __syncthreads();
    if (tid < 64) {
      lstate[tid] = lstate[tid] * alpha_s[tid] + rowsum2[tid * 2] + rowsum2[tid * 2 + 1];
      mstate[tid] = mnew_s[tid];
    }
    __syncthreads();

    // rescale accumulators, then out += P @ V  (V^T source -> K-contiguous)
#pragma unroll
    for (int nt = 0; nt < 4; ++nt)
#pragma unroll
      for (int r = 0; r < 8; ++r)
        oacc[nt][r] *= alpha_s[r0 + r + 8 * halfl];
#pragma unroll
    for (int kc = 0; kc < 2; ++kc) {
      v16bf af = load_a_frag(P + r0 * 64, 64, kc * 32, lane);
#pragma unroll
      for (int nt = 0; nt < 4; ++nt) {
        v16bf bt = load_bT_frag(&VsT[bank][(nt * 16) * 64], 64, kc * 32, lane);
        oacc[nt] = wmma_bf16(af, bt, oacc[nt]);
      }
    }
    __syncthreads();   // protect bank^1 tiles + P against next iteration
  }

  float* outp = out + ((size_t)b * XT + i0) * XD + h * XDH;
#pragma unroll
  for (int nt = 0; nt < 4; ++nt)
#pragma unroll
    for (int r = 0; r < 8; ++r) {
      const int m = r0 + r + 8 * halfl;
      outp[(size_t)m * XD + nt * 16 + l16] = oacc[nt][r] / lstate[m];
    }
}

// --------------------------------- launch ----------------------------------

extern "C" void kernel_launch(void* const* d_in, const int* in_sizes, int n_in,
                              void* d_out, int out_size, void* d_ws, size_t ws_size,
                              hipStream_t stream) {
  const float* x   = (const float*)d_in[0];
  const float* mem = (const float*)d_in[1];
  const float* Wq  = (const float*)d_in[2];
  const float* bq  = (const float*)d_in[3];
  const float* Wk  = (const float*)d_in[4];
  const float* bk  = (const float*)d_in[5];
  const float* Wv  = (const float*)d_in[6];
  const float* bv  = (const float*)d_in[7];
  const float* Wo  = (const float*)d_in[8];
  const float* bo  = (const float*)d_in[9];
  const float* Wr  = (const float*)d_in[10];
  const float* u   = (const float*)d_in[11];
  const float* v   = (const float*)d_in[12];
  const float* g1  = (const float*)d_in[13];
  const float* be1 = (const float*)d_in[14];
  const float* g2  = (const float*)d_in[15];
  const float* be2 = (const float*)d_in[16];
  const float* W1  = (const float*)d_in[17];
  const float* b1  = (const float*)d_in[18];
  const float* W2  = (const float*)d_in[19];
  const float* b2  = (const float*)d_in[20];
  const int* is_causal = (const int*)d_in[21];

  const size_t NTD = (size_t)XB * XT * XD;  // 8,388,608
  const size_t NKD = (size_t)XB * XK * XD;  // 16,777,216
  const size_t NRD = (size_t)XK * XD;       // 4,194,304
  const size_t NTF = (size_t)XB * XT * XF;  // 33,554,432

  char* p = (char*)d_ws;
  auto alloc = [&](size_t bytes) -> char* {
    char* r = p;
    p += (bytes + 255) & ~(size_t)255;
    return r;
  };

  bf16_t* xbf   = (bf16_t*)alloc(NTD * 2);
  bf16_t* cbf   = (bf16_t*)alloc(NKD * 2);
  bf16_t* posbf = (bf16_t*)alloc(NRD * 2);
  bf16_t* wqT   = (bf16_t*)alloc((size_t)XD * XD * 2);   // all weights stored N x K
  bf16_t* wkT   = (bf16_t*)alloc((size_t)XD * XD * 2);
  bf16_t* wvT   = (bf16_t*)alloc((size_t)XD * XD * 2);
  bf16_t* woT   = (bf16_t*)alloc((size_t)XD * XD * 2);
  bf16_t* wrT   = (bf16_t*)alloc((size_t)XD * XD * 2);
  bf16_t* w1T   = (bf16_t*)alloc((size_t)XD * XF * 2);   // [F][D]
  bf16_t* w2T   = (bf16_t*)alloc((size_t)XF * XD * 2);   // [D][F]
  float*  qf    = (float*)alloc(NTD * 4);
  float*  kf    = (float*)alloc(NKD * 4);  // kf..vf contiguous (aliased by ff)
  float*  vf    = (float*)alloc(NKD * 4);
  float*  rf    = (float*)alloc(NRD * 4);
  bf16_t* qubf  = (bf16_t*)alloc(NTD * 2); // qubf..kbh contiguous (aliased by fbf)
  bf16_t* qvbf  = (bf16_t*)alloc(NTD * 2);
  bf16_t* kbh   = (bf16_t*)alloc(NKD * 2);
  bf16_t* vbhT  = (bf16_t*)alloc(NKD * 2);
  bf16_t* rbh   = (bf16_t*)alloc(NRD * 2);
  float*  attnf = (float*)alloc(NTD * 4);
  float*  hf    = (float*)alloc(NTD * 4);
  // time-disjoint aliases (producers dead by the time these are written)
  float*  af   = qf;              // attn output projection result
  bf16_t* aobf = (bf16_t*)rf;     // attn output bf16
  bf16_t* hbf  = xbf;             // post-LN1 bf16
  float*  ff   = kf;              // FFN hidden f32 (kf+vf = 128M)
  bf16_t* fbf  = qubf;            // FFN hidden bf16 (qubf+qvbf+kbh = 64M)
  float*  f2f  = (float*)vbhT;    // FFN out f32

  auto g1d = [](size_t n) { return dim3((unsigned)((n + 255) / 256)); };

  // 1) precision conversion / transposed-weight prep
  xl_cvt_f32_bf16<<<g1d(NTD), 256, 0, stream>>>(x, xbf, NTD);
  xl_concat_cvt<<<g1d(NKD), 256, 0, stream>>>(mem, x, cbf);
  xl_posemb<<<g1d(NRD), 256, 0, stream>>>(posbf);
  xl_cvt_T<<<g1d((size_t)XD * XD), 256, 0, stream>>>(Wq, wqT, XD, XD);
  xl_cvt_T<<<g1d((size_t)XD * XD), 256, 0, stream>>>(Wk, wkT, XD, XD);
  xl_cvt_T<<<g1d((size_t)XD * XD), 256, 0, stream>>>(Wv, wvT, XD, XD);
  xl_cvt_T<<<g1d((size_t)XD * XD), 256, 0, stream>>>(Wo, woT, XD, XD);
  xl_cvt_T<<<g1d((size_t)XD * XD), 256, 0, stream>>>(Wr, wrT, XD, XD);
  xl_cvt_T<<<g1d((size_t)XD * XF), 256, 0, stream>>>(W1, w1T, XD, XF);
  xl_cvt_T<<<g1d((size_t)XF * XD), 256, 0, stream>>>(W2, w2T, XF, XD);

  // 2) projections
  xl_gemm_bf16<<<dim3(XD / 64, (XB * XT) / 64), 128, 0, stream>>>(
      xbf, wqT, bq, qf, XD, XD, 0);
  xl_gemm_bf16<<<dim3(XD / 64, (XB * XK) / 64), 128, 0, stream>>>(
      cbf, wkT, bk, kf, XD, XD, 0);
  xl_gemm_bf16<<<dim3(XD / 64, (XB * XK) / 64), 128, 0, stream>>>(
      cbf, wvT, bv, vf, XD, XD, 0);
  xl_gemm_bf16<<<dim3(XD / 64, XK / 64), 128, 0, stream>>>(
      posbf, wrT, (const float*)nullptr, rf, XD, XD, 0);

  // 3) head layout repack
  xl_prep_q<<<g1d(NTD), 256, 0, stream>>>(qf, u, v, qubf, qvbf);
  xl_prep_kv<<<g1d(NKD), 256, 0, stream>>>(kf, kbh);
  xl_prep_vT<<<g1d(NKD), 256, 0, stream>>>(vf, vbhT);
  xl_prep_r<<<g1d(NRD), 256, 0, stream>>>(rf, rbh);

  // 4) streaming attention with online softmax
  xl_flash_attn<<<dim3(XT / 64, XH, XB), 128, 0, stream>>>(
      qubf, qvbf, kbh, vbhT, rbh, is_causal, attnf);

  // 5) output projection + residual LN
  xl_cvt_f32_bf16<<<g1d(NTD), 256, 0, stream>>>(attnf, aobf, NTD);
  xl_gemm_bf16<<<dim3(XD / 64, (XB * XT) / 64), 128, 0, stream>>>(
      aobf, woT, bo, af, XD, XD, 0);
  xl_ln_res<<<dim3(XB * XT), 256, 0, stream>>>(af, x, g1, be1, hf);

  // 6) FFN
  xl_cvt_f32_bf16<<<g1d(NTD), 256, 0, stream>>>(hf, hbf, NTD);
  xl_gemm_bf16<<<dim3(XF / 64, (XB * XT) / 64), 128, 0, stream>>>(
      hbf, w1T, b1, ff, XF, XD, 1);
  xl_cvt_f32_bf16<<<g1d(NTF), 256, 0, stream>>>(ff, fbf, NTF);
  xl_gemm_bf16<<<dim3(XD / 64, (XB * XT) / 64), 128, 0, stream>>>(
      fbf, w2T, b2, f2f, XD, XF, 0);
  xl_ln_res<<<dim3(XB * XT), 256, 0, stream>>>(f2f, hf, g2, be2, (float*)d_out);
}
